// ErrorNeuronsCore_63273458205366
// MI455X (gfx1250) — matrix operand order repaired and checked by reading, not compiled, not run on hardware
//
#include <hip/hip_runtime.h>

// ---- problem constants (match reference) ----
#define Bb   2
#define Cc   3
#define Hh   128
#define Ww   128
#define Tt   (Hh * Ww)
#define Dd   147          // C * 7*7
#define DPAD 160          // padded K for 16x16x32 f16 WMMA (5 K-steps)

typedef _Float16 half8 __attribute__((ext_vector_type(8)));
typedef _Float16 v16h  __attribute__((ext_vector_type(16)));
typedef float    v8f   __attribute__((ext_vector_type(8)));

// -------------------------------------------------------------------------
// Kernel 1: per-pixel 7x7 LayerNorm features (f16) + feature L2 norms (f32).
//   Padding zeros ARE part of the 49-element mean/var (torch F.unfold pads
//   with zeros, then mean over k*k includes them). All taps loaded with
//   clamped addresses (unconditional loads) and masked arithmetically.
// -------------------------------------------------------------------------
__global__ void __launch_bounds__(256)
feat_kernel(const float* __restrict__ img,
            _Float16* __restrict__ F,
            float* __restrict__ norms)
{
    int idx = blockIdx.x * blockDim.x + threadIdx.x;   // [0, B*T)
    int b = idx >> 14;            // / T (T = 16384)
    int p = idx & (Tt - 1);
    int y = p >> 7;
    int x = p & 127;

    _Float16* frow = F + (size_t)idx * DPAD;
    float normsq = 0.f;

    for (int c = 0; c < Cc; ++c) {
        const float* ch = img + (size_t)(b * Cc + c) * Tt;
        float vals[49];
        float s1 = 0.f, s2 = 0.f;
#pragma unroll
        for (int ky = 0; ky < 7; ++ky) {
            int yy  = y + ky - 3;
            int yyc = yy < 0 ? 0 : (yy > Hh - 1 ? Hh - 1 : yy);
            float ymask = (yy >= 0 && yy < Hh) ? 1.f : 0.f;
#pragma unroll
            for (int kx = 0; kx < 7; ++kx) {
                int xx  = x + kx - 3;
                int xxc = xx < 0 ? 0 : (xx > Ww - 1 ? Ww - 1 : xx);
                float m = (xx >= 0 && xx < Ww) ? ymask : 0.f;
                float v = ch[yyc * Ww + xxc] * m;      // unconditional cached load
                vals[ky * 7 + kx] = v;
                s1 += v;
                s2 += v * v;
            }
        }
        float mu  = s1 * (1.f / 49.f);
        float var = s2 * (1.f / 49.f) - mu * mu;
        float inv = 1.f / sqrtf(var + 1e-5f);
#pragma unroll
        for (int k = 0; k < 49; ++k) {
            float f = (vals[k] - mu) * inv + 2.f;
            normsq += f * f;
            frow[c * 49 + k] = (_Float16)f;
        }
    }
#pragma unroll
    for (int d = Dd; d < DPAD; ++d) frow[d] = (_Float16)0.f;   // K padding -> 0
    norms[idx] = sqrtf(normsq);
}

// -------------------------------------------------------------------------
// Kernel 2: one wave32 per (b, row, 16-col tile).
//   A = features of 16 target pixels [16 x 160] (resident in 40 VGPRs).
//   For each dr in [-2,2]: two 16x16 Gram tiles vs candidate cols x0-2 and
//   x0+14 via v_wmma_f32_16x16x32_f16. All 10 B chunks are preloaded into
//   distinct register groups (sched_barrier keeps them hoisted) so the
//   WMMAs run with partial loadcnt waits instead of full serialization.
//   Neighbour dots are diagonals j-i = dc+2 in {0..4}; extracted via LDS.
// -------------------------------------------------------------------------
__global__ void __launch_bounds__(32)
sal_kernel(const _Float16* __restrict__ F,
           const float* __restrict__ norms,
           float* __restrict__ out)
{
    __shared__ float G0[16 * 16];
    __shared__ float G1[16 * 16];

    const int lane = threadIdx.x;           // 0..31, wave32
    const int bid  = blockIdx.x;
    const int b    = bid >> 10;             // / (128 rows * 8 tiles)
    const int rem  = bid & 1023;
    const int r    = rem >> 3;
    const int x0   = (rem & 7) << 4;

    const int  li = lane & 15;
    const bool lo = lane < 16;

    // ---- load A (16 targets x K). A layout used here (same K order as B):
    //      lane L holds row M=L&15, contiguous K in {kb..kb+15}, kb = lo?0:16
    v16h a[5];
    {
        const _Float16* row = F + (size_t)(b * Tt + r * Ww + x0 + li) * DPAD;
        int kb = lo ? 0 : 16;
#pragma unroll
        for (int i = 0; i < 5; ++i) {
            half8 h0 = *(const half8*)(row + i * 32 + kb);
            half8 h1 = *(const half8*)(row + i * 32 + kb + 8);
            v16h av;
#pragma unroll
            for (int e = 0; e < 8; ++e) { av[e] = h0[e]; av[8 + e] = h1[e]; }
            a[i] = av;
        }
    }

    const float n_i = norms[b * Tt + r * Ww + x0 + li];
    float acc = 0.f, cnt = 0.f;

    // B layout used here (same K order as A): lane holds col N=L&15, contiguous K {0..15} or
    // {16..31} -> each chunk is one 32-byte-aligned v16h load.
    const int kbB = lo ? 0 : 16;

    // per-lane candidate column offsets (dr-independent)
    int colA = x0 - 2 + li;
    colA = colA < 0 ? 0 : (colA > Ww - 1 ? Ww - 1 : colA);    // tile0 cols
    int colB = x0 + 14 + li;
    colB = colB > Ww - 1 ? Ww - 1 : colB;                     // tile1 cols

#pragma unroll 1
    for (int dr = -2; dr <= 2; ++dr) {
        int rr  = r + dr;
        int rrc = rr < 0 ? 0 : (rr > Hh - 1 ? Hh - 1 : rr);

        const _Float16* rowB0 = F + (size_t)(b * Tt + rrc * Ww + colA) * DPAD + kbB;
        const _Float16* rowB1 = F + (size_t)(b * Tt + rrc * Ww + colB) * DPAD + kbB;

        // ---- preload ALL B chunks for both Gram tiles (10 x v16h live) ----
        v16h b0v[5], b1v[5];
#pragma unroll
        for (int i = 0; i < 5; ++i) {
            b0v[i] = *(const v16h*)(rowB0 + i * 32);
            b1v[i] = *(const v16h*)(rowB1 + i * 32);
        }
        // keep the loads hoisted above the WMMA block: with all 10 operands
        // simultaneously live, RA must use distinct register groups and the
        // waits before each WMMA become partial (overlap loads with XDL).
        __builtin_amdgcn_sched_barrier(0);

        v8f c0 = {}, c1 = {};
#pragma unroll
        for (int i = 0; i < 5; ++i) {
            c0 = __builtin_amdgcn_wmma_f32_16x16x32_f16(
                    false, a[i], false, b0v[i], (short)0, c0, false, false);
            c1 = __builtin_amdgcn_wmma_f32_16x16x32_f16(
                    false, a[i], false, b1v[i], (short)0, c1, false, false);
        }

        // spill C/D tiles: lane holds column N=li, rows M = v + (lo?0:8)
#pragma unroll
        for (int v = 0; v < 8; ++v) {
            int M = v + (lo ? 0 : 8);
            G0[M * 16 + li] = c0[v];
            G1[M * 16 + li] = c1[v];
        }
        __syncthreads();

        if (rr >= 0 && rr < Hh) {   // uniform branch (depends on r, dr only)
#pragma unroll
            for (int dc = -2; dc <= 2; ++dc) {
                if (dr == 0 && dc == 0) continue;       // center removed
                int nx = x0 + li + dc;
                bool in = (nx >= 0) && (nx < Ww);
                int nxc = nx < 0 ? 0 : (nx > Ww - 1 ? Ww - 1 : nx);
                int j = li + dc + 2;                    // diagonal index
                float dot = (j < 16) ? G0[li * 16 + j] : G1[li * 16 + (j - 16)];
                float nn  = norms[b * Tt + rr * Ww + nxc];
                float dnm = fmaxf(n_i * nn, 1e-8f);
                float dst = (1.f - dot / dnm) * 0.5f;
                if (in) { acc += dst; cnt += 1.f; }
            }
        }
        __syncthreads();
    }

    if (lo) out[b * Tt + r * Ww + x0 + li] = acc / cnt;
}

// -------------------------------------------------------------------------
extern "C" void kernel_launch(void* const* d_in, const int* in_sizes, int n_in,
                              void* d_out, int out_size, void* d_ws, size_t ws_size,
                              hipStream_t stream)
{
    const float* img = (const float*)d_in[0];          // [2,3,128,128] f32
    _Float16* F     = (_Float16*)d_ws;                 // [B*T, 160] f16  (10.5 MB)
    float*    norms = (float*)((char*)d_ws + (size_t)Bb * Tt * DPAD * sizeof(_Float16));
    float*    out   = (float*)d_out;                   // [2,128,128] f32

    feat_kernel<<<(Bb * Tt) / 256, 256, 0, stream>>>(img, F, norms);
    sal_kernel<<<Bb * Hh * (Ww / 16), 32, 0, stream>>>(F, norms, out);
}
